// SMPLTorchModel_16028817949189
// MI455X (gfx1250) — matrix operand order repaired and checked by reading, not compiled, not run on hardware
//
#include <hip/hip_runtime.h>
#include <hip/hip_bf16.h>
#include <math.h>

// ---- problem constants ----
#define BATCH 512
#define NVERT 6890
#define NJ    24
#define N3    (NVERT*3)      // 20670 flattened vertex-coord columns
#define KP    224            // padded K: 10 (betas) + 207 (lrotmin) + 7 pad
#define NP    20672          // N3 padded to multiple of 16 (1292 tiles)
#define MT    (BATCH/16)     // 32 M tiles
#define NT    (NP/16)        // 1292 N tiles

typedef _Float16 v16h __attribute__((ext_vector_type(16)));
typedef _Float16 v8h  __attribute__((ext_vector_type(8)));
typedef float    v8f  __attribute__((ext_vector_type(8)));

__device__ __constant__ int kPar[NJ] = {-1,0,0,0,1,2,3,4,5,6,7,8,9,9,9,12,13,14,16,17,18,19,20,21};

// ---------------------------------------------------------------------------
// Kernel 1: pack B matrix [NP][KP] f16 = [shapedirs(10) | posedirs(207) | 0]
// row n = v*3+c (K-contiguous so the WMMA B-fragment is one contiguous load)
// ---------------------------------------------------------------------------
__global__ __launch_bounds__(256) void smpl_pack_b(
    const float* __restrict__ shapedirs,   // [N3][10]
    const float* __restrict__ posedirs,    // [N3][207]
    _Float16* __restrict__ Bf16)           // [NP][KP]
{
  int idx = blockIdx.x * 256 + threadIdx.x;
  if (idx >= NP * KP) return;
  int n = idx / KP, k = idx % KP;
  float v = 0.0f;
  if (n < N3) {
    if (k < 10)        v = shapedirs[n * 10 + k];
    else if (k < 217)  v = posedirs[n * 207 + (k - 10)];
  }
  Bf16[idx] = (_Float16)v;
}

// ---------------------------------------------------------------------------
// Kernel 2: Q[(j*3+c)*11 + k] = sum_v Jreg[j][v] * (k<10 ? shapedirs[v,c,k]
//                                                        : v_template[v,c])
// ---------------------------------------------------------------------------
__global__ __launch_bounds__(256) void smpl_q(
    const float* __restrict__ Jreg,        // [NJ][NVERT]
    const float* __restrict__ shapedirs,   // [N3][10]
    const float* __restrict__ vtpl,        // [N3]
    float* __restrict__ Q)                 // [NJ*3*11]
{
  int o  = blockIdx.x;                     // 0..791
  int k  = o % 11;
  int jc = o / 11;
  int c  = jc % 3, j = jc / 3;
  const float* jr = Jreg + j * NVERT;
  float s = 0.0f;
  if (k < 10) {
    for (int v = threadIdx.x; v < NVERT; v += 256)
      s += jr[v] * shapedirs[(v * 3 + c) * 10 + k];
  } else {
    for (int v = threadIdx.x; v < NVERT; v += 256)
      s += jr[v] * vtpl[v * 3 + c];
  }
  __shared__ float red[256];
  red[threadIdx.x] = s;
  __syncthreads();
  for (int off = 128; off > 0; off >>= 1) {
    if (threadIdx.x < off) red[threadIdx.x] += red[threadIdx.x + off];
    __syncthreads();
  }
  if (threadIdx.x == 0) Q[o] = red[0];
}

// ---------------------------------------------------------------------------
// Kernel 3: per-batch rodrigues + joints + kinematic chain.
// Writes G' rows 0..2 ([BATCH][NJ][12]) and f16 A-row [betas|lrotmin|0].
// One wave (32 threads) per batch element.
// ---------------------------------------------------------------------------
__global__ __launch_bounds__(32) void smpl_chain(
    const float* __restrict__ betas,       // [BATCH][10]
    const float* __restrict__ pose,        // [BATCH][72]
    const float* __restrict__ Q,           // [792]
    float* __restrict__ Gout,              // [BATCH][NJ][12]
    _Float16* __restrict__ Af16)           // [BATCH][KP]
{
  __shared__ float sQ[792];
  __shared__ float sb[10];
  __shared__ float sJ[NJ][3];
  __shared__ float sR[NJ][9];
  __shared__ float sA[NJ][16];
  __shared__ float sG[NJ][16];
  const int b = blockIdx.x;
  const int t = threadIdx.x;

  for (int i = t; i < 792; i += 32) sQ[i] = Q[i];
  if (t < 10) sb[t] = betas[b * 10 + t];
  __syncthreads();

  if (t < NJ) {
    // joint positions from shaped template
    for (int c = 0; c < 3; ++c) {
      const float* q = sQ + (t * 3 + c) * 11;
      float s = q[10];
      #pragma unroll
      for (int k = 0; k < 10; ++k) s += sb[k] * q[k];
      sJ[t][c] = s;
    }
    // rodrigues
    float rx = pose[b * 72 + t * 3 + 0];
    float ry = pose[b * 72 + t * 3 + 1];
    float rz = pose[b * 72 + t * 3 + 2];
    float th = sqrtf(rx * rx + ry * ry + rz * rz) + 1e-8f;
    float iv = 1.0f / th;
    float x = rx * iv, y = ry * iv, z = rz * iv;
    float c0 = cosf(th), s0 = sinf(th), om = 1.0f - c0;
    sR[t][0] = c0 + om * x * x;  sR[t][1] = om * x * y - s0 * z;  sR[t][2] = om * x * z + s0 * y;
    sR[t][3] = om * y * x + s0 * z;  sR[t][4] = c0 + om * y * y;  sR[t][5] = om * y * z - s0 * x;
    sR[t][6] = om * z * x - s0 * y;  sR[t][7] = om * z * y + s0 * x;  sR[t][8] = c0 + om * z * z;
  }
  __syncthreads();

  // A row in f16: [betas(10) | lrotmin(207) | zeros(7)]
  _Float16* arow = Af16 + (size_t)b * KP;
  if (t < 10) arow[t] = (_Float16)sb[t];
  for (int p = t; p < 207; p += 32) {
    int j = 1 + p / 9, e = p % 9;
    float diag = (e == 0 || e == 4 || e == 8) ? 1.0f : 0.0f;
    arow[10 + p] = (_Float16)(sR[j][e] - diag);
  }
  if (t < 7) arow[217 + t] = (_Float16)0.0f;

  // local transforms
  if (t < NJ) {
    int p = kPar[t];
    float tx, ty, tz;
    if (t == 0) { tx = sJ[0][0]; ty = sJ[0][1]; tz = sJ[0][2]; }
    else { tx = sJ[t][0] - sJ[p][0]; ty = sJ[t][1] - sJ[p][1]; tz = sJ[t][2] - sJ[p][2]; }
    sA[t][0] = sR[t][0]; sA[t][1] = sR[t][1]; sA[t][2]  = sR[t][2]; sA[t][3]  = tx;
    sA[t][4] = sR[t][3]; sA[t][5] = sR[t][4]; sA[t][6]  = sR[t][5]; sA[t][7]  = ty;
    sA[t][8] = sR[t][6]; sA[t][9] = sR[t][7]; sA[t][10] = sR[t][8]; sA[t][11] = tz;
    sA[t][12] = 0.0f; sA[t][13] = 0.0f; sA[t][14] = 0.0f; sA[t][15] = 1.0f;
  }
  __syncthreads();
  if (t < 16) sG[0][t] = sA[0][t];
  __syncthreads();
  for (int i = 1; i < NJ; ++i) {
    if (t < 16) {
      int r = t >> 2, cc = t & 3;
      int p = kPar[i];
      float s = 0.0f;
      #pragma unroll
      for (int y = 0; y < 4; ++y) s += sG[p][r * 4 + y] * sA[i][y * 4 + cc];
      sG[i][t] = s;
    }
    __syncthreads();
  }
  // subtract t = G * [J,0] from column 3; emit rows 0..2
  if (t < NJ) {
    float jx = sJ[t][0], jy = sJ[t][1], jz = sJ[t][2];
    float* go = Gout + ((size_t)b * NJ + t) * 12;
    #pragma unroll
    for (int r = 0; r < 3; ++r) {
      float g0 = sG[t][r * 4 + 0], g1 = sG[t][r * 4 + 1];
      float g2 = sG[t][r * 4 + 2], g3 = sG[t][r * 4 + 3];
      go[r * 4 + 0] = g0; go[r * 4 + 1] = g1; go[r * 4 + 2] = g2;
      go[r * 4 + 3] = g3 - (g0 * jx + g1 * jy + g2 * jz);
    }
  }
}

// ---------------------------------------------------------------------------
// Kernel 4: fused blend GEMM via WMMA.
// v_posed[b][n] = v_template[n] + sum_k A[b][k]*B[n][k], K=224, one 16x16
// tile per wave, 7 chained v_wmma_f32_16x16x32_f16.
// ---------------------------------------------------------------------------
__global__ __launch_bounds__(256) void smpl_gemm_wmma(
    const _Float16* __restrict__ Af16,     // [BATCH][KP]
    const _Float16* __restrict__ Bf16,     // [NP][KP]
    const float* __restrict__ vtpl,        // [N3]
    float* __restrict__ vposed)            // [BATCH][NP]
{
  const int lane = threadIdx.x & 31;
  const int wave = threadIdx.x >> 5;
  const int tile = blockIdx.x * 8 + wave;  // grid sized exactly: MT*NT tiles
  const int mt = tile & (MT - 1);          // M tile (batch/16)
  const int nt = tile / MT;                // N tile
  const int m0 = mt << 4;
  const int n0 = nt << 4;
  const int half = lane >> 4;              // lane group 0/1
  const int l16  = lane & 15;

  const int n = n0 + l16;                  // this lane's output column
  float vt = (n < N3) ? vtpl[n] : 0.0f;    // accumulator init = template
  v8f acc;
  #pragma unroll
  for (int i = 0; i < 8; ++i) acc[i] = vt;

  // A fragment: lane holds row m = l16; 16 contiguous K starting at half*16
  const _Float16* arow = Af16 + (size_t)(m0 + l16) * KP + (half << 4);
  // B fragment: lane holds col n; 16 contiguous K starting at half*16
  const _Float16* brow = Bf16 + (size_t)n * KP + (half << 4);

  #pragma unroll
  for (int kb = 0; kb < KP; kb += 32) {
    v16h a = *(const v16h*)(arow + kb);
    v16h bf = *(const v16h*)(brow + kb);
    acc = __builtin_amdgcn_wmma_f32_16x16x32_f16(
        /*neg_a=*/false, a, /*neg_b=*/false, bf,
        /*c_mod=*/(short)0, acc, /*reuse_a=*/false, /*reuse_b=*/false);
  }
  // D layout: VGPR r -> row m0 + (lane>>4)*8 + r, col n
  float* out = vposed + (size_t)(m0 + (half << 3)) * NP + n;
  #pragma unroll
  for (int r = 0; r < 8; ++r) out[(size_t)r * NP] = acc[r];
}

// ---------------------------------------------------------------------------
// Kernel 5: LBS skinning. One block per batch element, G' rows in LDS.
// result[b][v][c] = sum_j w[v][j]*(G'[b][j][c]·[vp,1]) + trans[b][c]
// ---------------------------------------------------------------------------
__global__ __launch_bounds__(256) void smpl_lbs(
    const float* __restrict__ vposed,      // [BATCH][NP]
    const float* __restrict__ Gws,         // [BATCH][NJ][12]
    const float* __restrict__ weights,     // [NVERT][NJ]
    const float* __restrict__ trans,       // [BATCH][3]
    float* __restrict__ result)            // [BATCH][NVERT][3]
{
  __shared__ float sG[NJ * 12];
  __shared__ float st[3];
  const int b = blockIdx.x, t = threadIdx.x;
  for (int i = t; i < NJ * 12; i += 256) sG[i] = Gws[(size_t)b * NJ * 12 + i];
  if (t < 3) st[t] = trans[b * 3 + t];
  __syncthreads();
  const float* vp = vposed + (size_t)b * NP;
  float* out = result + (size_t)b * N3;
  for (int v = t; v < NVERT; v += 256) {
    float T[12];
    #pragma unroll
    for (int e = 0; e < 12; ++e) T[e] = 0.0f;
    const float* w = weights + v * NJ;
    for (int j = 0; j < NJ; ++j) {
      float wj = w[j];
      const float* g = sG + j * 12;
      #pragma unroll
      for (int e = 0; e < 12; ++e) T[e] += wj * g[e];
    }
    float x = vp[v * 3 + 0], y = vp[v * 3 + 1], z = vp[v * 3 + 2];
    #pragma unroll
    for (int c = 0; c < 3; ++c)
      out[v * 3 + c] = T[c * 4 + 0] * x + T[c * 4 + 1] * y +
                       T[c * 4 + 2] * z + T[c * 4 + 3] + st[c];
  }
}

// ---------------------------------------------------------------------------
// Kernel 6: joints = joint_regressor @ result. One block per (b, j).
// ---------------------------------------------------------------------------
__global__ __launch_bounds__(256) void smpl_joints(
    const float* __restrict__ result,      // [BATCH][NVERT][3]
    const float* __restrict__ jreg,        // [NJ][NVERT]
    float* __restrict__ joints)            // [BATCH][NJ][3]
{
  const int o = blockIdx.x;                // b*NJ + j
  const int j = o % NJ, b = o / NJ;
  const int t = threadIdx.x;
  const float* jr  = jreg + (size_t)j * NVERT;
  const float* res = result + (size_t)b * N3;
  float s0 = 0.0f, s1 = 0.0f, s2 = 0.0f;
  for (int v = t; v < NVERT; v += 256) {
    float w = jr[v];
    s0 += w * res[v * 3 + 0];
    s1 += w * res[v * 3 + 1];
    s2 += w * res[v * 3 + 2];
  }
  __shared__ float r0[256], r1[256], r2[256];
  r0[t] = s0; r1[t] = s1; r2[t] = s2;
  __syncthreads();
  for (int off = 128; off > 0; off >>= 1) {
    if (t < off) { r0[t] += r0[t + off]; r1[t] += r1[t + off]; r2[t] += r2[t + off]; }
    __syncthreads();
  }
  if (t == 0) {
    joints[o * 3 + 0] = r0[0];
    joints[o * 3 + 1] = r1[0];
    joints[o * 3 + 2] = r2[0];
  }
}

// ---------------------------------------------------------------------------
extern "C" void kernel_launch(void* const* d_in, const int* in_sizes, int n_in,
                              void* d_out, int out_size, void* d_ws, size_t ws_size,
                              hipStream_t stream) {
  (void)in_sizes; (void)n_in; (void)out_size; (void)ws_size;
  const float* betas     = (const float*)d_in[0];
  const float* pose      = (const float*)d_in[1];
  const float* trans     = (const float*)d_in[2];
  const float* vtpl      = (const float*)d_in[3];
  const float* shapedirs = (const float*)d_in[4];
  const float* posedirs  = (const float*)d_in[5];
  const float* Jreg      = (const float*)d_in[6];
  const float* jreg2     = (const float*)d_in[7];
  const float* weights   = (const float*)d_in[8];

  float* result = (float*)d_out;
  float* joints = result + (size_t)BATCH * N3;

  // workspace carve-out (all offsets multiples of 512 bytes)
  char* ws = (char*)d_ws;
  _Float16* Af16 = (_Float16*)ws;  ws += (size_t)BATCH * KP * 2;     // 224 KiB
  _Float16* Bf16 = (_Float16*)ws;  ws += (size_t)NP * KP * 2;        // ~8.8 MiB
  float* Q       = (float*)ws;     ws += 1024 * 4;                    // 792 used
  float* Gws     = (float*)ws;     ws += (size_t)BATCH * NJ * 12 * 4; // 576 KiB
  float* vposed  = (float*)ws;                                        // ~40.4 MiB

  smpl_pack_b  <<<(NP * KP + 255) / 256, 256, 0, stream>>>(shapedirs, posedirs, Bf16);
  smpl_q       <<<NJ * 3 * 11, 256, 0, stream>>>(Jreg, shapedirs, vtpl, Q);
  smpl_chain   <<<BATCH, 32, 0, stream>>>(betas, pose, Q, Gws, Af16);
  smpl_gemm_wmma<<<(MT * NT) / 8, 256, 0, stream>>>(Af16, Bf16, vtpl, vposed);
  smpl_lbs     <<<BATCH, 256, 0, stream>>>(vposed, Gws, weights, trans, result);
  smpl_joints  <<<BATCH * NJ, 256, 0, stream>>>(result, jreg2, joints);
}
